// ESPNETMultiHeadedAttention_884763263182
// MI455X (gfx1250) — matrix operand order repaired and checked by reading, not compiled, not run on hardware
//
#include <hip/hip_runtime.h>

#define T_DIM 2048
#define BATCH 4
#define C_DIM 1024
#define H_DIM 16
#define DK    64
#define M_ROWS (T_DIM * BATCH)   // 8192
#define CLAMPV 1e8f

typedef _Float16 v16h  __attribute__((ext_vector_type(16)));
typedef _Float16 half8 __attribute__((ext_vector_type(8)));
typedef _Float16 half4 __attribute__((ext_vector_type(4)));
typedef float    v8f   __attribute__((ext_vector_type(8)));
typedef float    f4v   __attribute__((ext_vector_type(4)));

// ---------------------------------------------------------------------------
// Streaming f32 -> f16 down-convert (one pass per tensor; all reuse of the
// data afterwards is pure f16 loads, halving L2/HBM bytes in the GEMMs).
// ---------------------------------------------------------------------------
__global__ __launch_bounds__(256) void cvt_f32_to_f16(
    const float* __restrict__ src, _Float16* __restrict__ dst, int n4) {
  const int i = blockIdx.x * 256 + threadIdx.x;
  if (i < n4) {
    f4v x = ((const f4v*)src)[i];
    half4 y;
#pragma unroll
    for (int k = 0; k < 4; ++k) y[k] = (_Float16)x[k];
    ((half4*)dst)[i] = y;
  }
}

// ---------------------------------------------------------------------------
// Fragment builders (this file's contiguous k order, same for A and B, wave32)
// A (16x32 f16): lane L: M = L&15; elems 0..15 -> K = 2*kb..2*kb+15,
//                kb = (L>>4)*8
// B (32x16 f16): lane L: N = L&15; elems 0..15 -> K = kb2..kb2+15, kb2=(L>>4)*16
// C/D (16x16 f32): lane L: N = L&15; vgpr r -> M = r + 8*(L>>4)
// ---------------------------------------------------------------------------
__device__ __forceinline__ v16h a_frag_from_f16(const _Float16* rowbase, int kb) {
  half8 lo = *(const half8*)(rowbase + 2 * kb);
  half8 hv = *(const half8*)(rowbase + 2 * kb + 8);
  v16h a;
#pragma unroll
  for (int i = 0; i < 8; ++i) { a[i] = lo[i]; a[8 + i] = hv[i]; }
  return a;
}

// ---------------------------------------------------------------------------
// GEMM: out[m,n] = A[m,:] . W[n,:] + bias[n]   (torch Linear x @ W.T + b)
// A and W are f16. One wave computes a 32x64 output tile: 2 A fragments
// reused across 4 W fragments -> 8 WMMAs per 6KB of f16 loads.
// MODE 0: Q proj  -> f16 [B][H][T][DK], pre-scaled by 1/sqrt(DK)
// MODE 1: K proj  -> f16 [B][H][T][DK]
// MODE 2: V proj  -> f16 [B][H][DK][T]   (transposed for PV B-fragments)
// MODE 3: O proj  -> f32 [T][B][C] (== row-major m*C+n)
// ---------------------------------------------------------------------------
template <int MODE>
__global__ __launch_bounds__(32) void gemm32x64(
    const _Float16* __restrict__ A, const _Float16* __restrict__ Wh,
    const float* __restrict__ bias, void* __restrict__ outraw) {
  const int lane = threadIdx.x;
  const int l15  = lane & 15;
  const int hi   = lane >> 4;
  const int mbase = blockIdx.x * 32;
  const int nbase = blockIdx.y * 64;
  const int K = C_DIM;

  v8f acc[2][4] = {};
  const int kbA = hi * 8;
  const int kbB = hi * 16;

  for (int kc = 0; kc < K; kc += 32) {
    v16h afrag[2];
#pragma unroll
    for (int u = 0; u < 2; ++u) {
      const _Float16* ap =
          A + (size_t)(mbase + u * 16 + l15) * K + kc;
      afrag[u] = a_frag_from_f16(ap, kbA);
      if (kc + 32 < K) __builtin_prefetch(ap + 32, 0, 3);  // global_prefetch_b8
    }
#pragma unroll
    for (int g = 0; g < 4; ++g) {
      const _Float16* wp =
          Wh + (size_t)(nbase + g * 16 + l15) * K + kc + kbB;
      if (kc + 32 < K) __builtin_prefetch(wp + 32, 0, 3);
      v16h bfrag = *(const v16h*)wp;  // 16 contiguous K values for column N
      acc[0][g] = __builtin_amdgcn_wmma_f32_16x16x32_f16(
          false, afrag[0], false, bfrag, (short)0, acc[0][g], false, false);
      acc[1][g] = __builtin_amdgcn_wmma_f32_16x16x32_f16(
          false, afrag[1], false, bfrag, (short)0, acc[1][g], false, false);
    }
  }

#pragma unroll
  for (int u = 0; u < 2; ++u) {
#pragma unroll
    for (int g = 0; g < 4; ++g) {
      const int n = nbase + g * 16 + l15;
      const float bval = bias[n];
#pragma unroll
      for (int r = 0; r < 8; ++r) {
        const int m = mbase + u * 16 + r + hi * 8;
        float v = acc[u][g][r] + bval;
        if constexpr (MODE == 0 || MODE == 1) {
          const int t = m >> 2, b = m & 3;       // m = t*BATCH + b
          const int h = n >> 6, dk = n & 63;     // n = h*DK + dk
          if constexpr (MODE == 0) v *= 0.125f;  // 1/sqrt(64) folded into Q
          ((_Float16*)outraw)[((size_t)(b * H_DIM + h) * T_DIM + t) * DK + dk] =
              (_Float16)v;
        } else if constexpr (MODE == 2) {
          const int t = m >> 2, b = m & 3;
          const int h = n >> 6, dk = n & 63;
          ((_Float16*)outraw)[((size_t)(b * H_DIM + h) * DK + dk) * T_DIM + t] =
              (_Float16)v;
        } else {
          ((float*)outraw)[(size_t)m * C_DIM + n] = v;
        }
      }
    }
  }
}

// ---------------------------------------------------------------------------
// Flash attention: one wave per (b, h, 16-row query tile).
// Iterate keys 32 at a time: 4 score WMMAs + online softmax + 4 PV WMMAs.
// P is transposed C-layout -> A-layout through a 16x32 f16 LDS tile.
// ---------------------------------------------------------------------------
__global__ __launch_bounds__(32) void flash_attn(
    const _Float16* __restrict__ Q, const _Float16* __restrict__ Kh,
    const _Float16* __restrict__ VT, const unsigned char* __restrict__ kpm,
    _Float16* __restrict__ X) {
  __shared__ __align__(64) _Float16 plds[16 * 32];

  const int lane = threadIdx.x;
  const int l15  = lane & 15;
  const int hi   = lane >> 4;
  const int qt = blockIdx.x, h = blockIdx.y, b = blockIdx.z;
  const int bh = b * H_DIM + h;

  // Q A-fragments for both DK chunks (K = 0..31, 32..63)
  const _Float16* qrow = Q + ((size_t)bh * T_DIM + qt * 16 + l15) * DK;
  v16h aq[2];
#pragma unroll
  for (int c = 0; c < 2; ++c) aq[c] = a_frag_from_f16(qrow + c * 32, hi * 8);

  v8f oacc[4] = {};
  float mrun[8], lrun[8];
#pragma unroll
  for (int r = 0; r < 8; ++r) { mrun[r] = -1e30f; lrun[r] = 0.0f; }

  const unsigned char* kpmb = kpm + (size_t)b * T_DIM;
  const _Float16* kbase = Kh + (size_t)bh * T_DIM * DK;

  for (int j = 0; j < T_DIM; j += 32) {
    // ---- prefetch next key/value tiles one step ahead (global_prefetch_b8)
    if (j + 32 < T_DIM) {
      __builtin_prefetch(kbase + (size_t)(j + 32 + lane) * DK, 0, 3);
      __builtin_prefetch(VT + ((size_t)bh * DK + lane) * T_DIM + j + 32, 0, 3);
    }

    // ---- scores: two 16(q) x 16(key) tiles, K-dim = DK = 64 ----
    v8f sacc[2];
#pragma unroll
    for (int s = 0; s < 2; ++s) {
      const _Float16* krow = kbase + (size_t)(j + s * 16 + l15) * DK;
      v8f c = {};
#pragma unroll
      for (int ch = 0; ch < 2; ++ch) {
        v16h bk = *(const v16h*)(krow + ch * 32 + hi * 16);
        c = __builtin_amdgcn_wmma_f32_16x16x32_f16(
            false, aq[ch], false, bk, (short)0, c, false, false);
      }
      sacc[s] = c;
    }

    // ---- mask + clamp (masked -> -inf -> clip -> -1e8 per reference) ----
    const bool m0 = kpmb[j + l15] != 0;
    const bool m1 = kpmb[j + 16 + l15] != 0;
    float s0[8], s1[8];
#pragma unroll
    for (int r = 0; r < 8; ++r) {
      float a = fminf(fmaxf(sacc[0][r], -CLAMPV), CLAMPV);
      float c = fminf(fmaxf(sacc[1][r], -CLAMPV), CLAMPV);
      s0[r] = m0 ? -CLAMPV : a;
      s1[r] = m1 ? -CLAMPV : c;
    }

    // ---- online softmax (row = vgpr index; keys spread over 16 lanes) ----
    float alpha[8];
#pragma unroll
    for (int r = 0; r < 8; ++r) {
      float v = fmaxf(s0[r], s1[r]);
#pragma unroll
      for (int d = 1; d < 16; d <<= 1) v = fmaxf(v, __shfl_xor(v, d, 32));
      const float mn = fmaxf(mrun[r], v);
      alpha[r] = __expf(mrun[r] - mn);
      mrun[r] = mn;
      const float p0 = __expf(s0[r] - mn);
      const float p1 = __expf(s1[r] - mn);
      s0[r] = p0;
      s1[r] = p1;
      float rs = p0 + p1;
#pragma unroll
      for (int d = 1; d < 16; d <<= 1) rs += __shfl_xor(rs, d, 32);
      lrun[r] = lrun[r] * alpha[r] + rs;
    }
#pragma unroll
    for (int g = 0; g < 4; ++g)
#pragma unroll
      for (int r = 0; r < 8; ++r) oacc[g][r] *= alpha[r];

    // ---- transpose P through LDS: C-layout (lane=key) -> A-layout (lane=q)
#pragma unroll
    for (int r = 0; r < 8; ++r) {
      const int row = r + hi * 8;
      plds[row * 32 + l15]      = (_Float16)s0[r];
      plds[row * 32 + 16 + l15] = (_Float16)s1[r];
    }
    asm volatile("s_wait_dscnt 0" ::: "memory");  // cross-lane LDS visibility
    const v16h pa = a_frag_from_f16(&plds[l15 * 32], hi * 8);

    // ---- P(16x32) @ V(32x16) per 16-wide dk group ----
#pragma unroll
    for (int g = 0; g < 4; ++g) {
      const _Float16* vrow =
          VT + ((size_t)bh * DK + g * 16 + l15) * T_DIM + j + hi * 16;
      v16h bv = *(const v16h*)vrow;
      oacc[g] = __builtin_amdgcn_wmma_f32_16x16x32_f16(
          false, pa, false, bv, (short)0, oacc[g], false, false);
    }
  }

  // ---- finalize: divide by row sums, store f16 x in [T][B][C] order ----
#pragma unroll
  for (int g = 0; g < 4; ++g) {
#pragma unroll
    for (int r = 0; r < 8; ++r) {
      const float v = oacc[g][r] / lrun[r];
      const int q = qt * 16 + r + hi * 8;
      X[((size_t)q * BATCH + b) * C_DIM + h * DK + g * 16 + l15] = (_Float16)v;
    }
  }
}

// ---------------------------------------------------------------------------
extern "C" void kernel_launch(void* const* d_in, const int* in_sizes, int n_in,
                              void* d_out, int out_size, void* d_ws,
                              size_t ws_size, hipStream_t stream) {
  (void)in_sizes; (void)n_in; (void)out_size; (void)ws_size;
  const float* query = (const float*)d_in[0];
  const float* key   = (const float*)d_in[1];
  const float* value = (const float*)d_in[2];
  const unsigned char* kpm = (const unsigned char*)d_in[3];  // jax bool, 1B
  const float* Wq = (const float*)d_in[4];
  const float* bq = (const float*)d_in[5];
  const float* Wk = (const float*)d_in[6];
  const float* bk = (const float*)d_in[7];
  const float* Wv = (const float*)d_in[8];
  const float* bv = (const float*)d_in[9];
  const float* Wo = (const float*)d_in[10];
  const float* bo = (const float*)d_in[11];

  // Workspace layout (all f16):
  //   4 x seg (16MB): wsQ, wsK, wsVT, wsX
  //   3 x inseg (16MB): query/key/value f16 copies
  //   4 x wseg (2MB): Wq/Wk/Wv/Wo f16 copies
  char* ws = (char*)d_ws;
  const size_t seg   = (size_t)BATCH * H_DIM * T_DIM * DK * sizeof(_Float16);
  const size_t inseg = (size_t)M_ROWS * C_DIM * sizeof(_Float16);
  const size_t wseg  = (size_t)C_DIM * C_DIM * sizeof(_Float16);
  _Float16* wsQ  = (_Float16*)(ws);
  _Float16* wsK  = (_Float16*)(ws + seg);
  _Float16* wsVT = (_Float16*)(ws + 2 * seg);
  _Float16* wsX  = (_Float16*)(ws + 3 * seg);
  _Float16* hq   = (_Float16*)(ws + 4 * seg);
  _Float16* hk   = (_Float16*)(ws + 4 * seg + inseg);
  _Float16* hv   = (_Float16*)(ws + 4 * seg + 2 * inseg);
  _Float16* hWq  = (_Float16*)(ws + 4 * seg + 3 * inseg);
  _Float16* hWk  = (_Float16*)(ws + 4 * seg + 3 * inseg + wseg);
  _Float16* hWv  = (_Float16*)(ws + 4 * seg + 3 * inseg + 2 * wseg);
  _Float16* hWo  = (_Float16*)(ws + 4 * seg + 3 * inseg + 3 * wseg);

  // ---- one-time f32 -> f16 down-converts (streaming, HBM-bound) ----
  const int nIn4 = (M_ROWS * C_DIM) / 4;   // 2M vec4
  const int nW4  = (C_DIM * C_DIM) / 4;    // 256K vec4
  const dim3 cblk(256, 1, 1);
  hipLaunchKernelGGL(cvt_f32_to_f16, dim3((nIn4 + 255) / 256), cblk, 0, stream,
                     query, hq, nIn4);
  hipLaunchKernelGGL(cvt_f32_to_f16, dim3((nIn4 + 255) / 256), cblk, 0, stream,
                     key, hk, nIn4);
  hipLaunchKernelGGL(cvt_f32_to_f16, dim3((nIn4 + 255) / 256), cblk, 0, stream,
                     value, hv, nIn4);
  hipLaunchKernelGGL(cvt_f32_to_f16, dim3((nW4 + 255) / 256), cblk, 0, stream,
                     Wq, hWq, nW4);
  hipLaunchKernelGGL(cvt_f32_to_f16, dim3((nW4 + 255) / 256), cblk, 0, stream,
                     Wk, hWk, nW4);
  hipLaunchKernelGGL(cvt_f32_to_f16, dim3((nW4 + 255) / 256), cblk, 0, stream,
                     Wv, hWv, nW4);
  hipLaunchKernelGGL(cvt_f32_to_f16, dim3((nW4 + 255) / 256), cblk, 0, stream,
                     Wo, hWo, nW4);

  // ---- WMMA GEMM projections (pure f16 fragment loads) ----
  const dim3 blk(32, 1, 1);
  const dim3 ggrid(M_ROWS / 32, C_DIM / 64, 1);
  hipLaunchKernelGGL((gemm32x64<0>), ggrid, blk, 0, stream, hq, hWq, bq,
                     (void*)wsQ);
  hipLaunchKernelGGL((gemm32x64<1>), ggrid, blk, 0, stream, hk, hWk, bk,
                     (void*)wsK);
  hipLaunchKernelGGL((gemm32x64<2>), ggrid, blk, 0, stream, hv, hWv, bv,
                     (void*)wsVT);

  // ---- WMMA flash attention ----
  const dim3 agrid(T_DIM / 16, H_DIM, BATCH);
  hipLaunchKernelGGL(flash_attn, agrid, blk, 0, stream, wsQ, wsK, wsVT, kpm,
                     wsX);

  // ---- WMMA output projection -> f32 (T,B,C) ----
  hipLaunchKernelGGL((gemm32x64<3>), ggrid, blk, 0, stream, wsX, hWo, bo,
                     d_out);
}
